// SelfAttention_14018773254573
// MI455X (gfx1250) — matrix operand order repaired and checked by reading, not compiled, not run on hardware
//
#include <hip/hip_runtime.h>
#include <hip/hip_bf16.h>

typedef _Float16 h8   __attribute__((ext_vector_type(8)));
typedef _Float16 v16h __attribute__((ext_vector_type(16)));
typedef float    v8f  __attribute__((ext_vector_type(8)));
typedef float    f4   __attribute__((ext_vector_type(4)));
typedef int      v4i  __attribute__((ext_vector_type(4)));

#define NEG_INF (-__builtin_inff())

#if defined(__AMDGCN__) && __has_builtin(__builtin_amdgcn_global_load_async_to_lds_b128) && \
    __has_builtin(__builtin_amdgcn_s_wait_asynccnt)
#define HAVE_ASYNC 1
#else
#define HAVE_ASYNC 0
#endif

#define AS_GLOBAL __attribute__((address_space(1)))
#define AS_LOCAL  __attribute__((address_space(3)))

// Copy 16 bytes global->LDS. Async path uses the CDNA5 ASYNCcnt-tracked DMA
// (global_load_async_to_lds_b128); fallback is a plain VGPR round-trip.
static __device__ __forceinline__ void cp16_g2l(const void* g, void* l) {
#if HAVE_ASYNC
    __builtin_amdgcn_global_load_async_to_lds_b128(
        (AS_GLOBAL v4i*)g, (AS_LOCAL v4i*)l, 0, 0);
#else
    *(h8*)l = *(const h8*)g;
#endif
}

static __device__ __forceinline__ v16h ld_frag16(const _Float16* p0, const _Float16* p1) {
    h8 lo = *(const h8*)p0;
    h8 hi = *(const h8*)p1;
    return __builtin_shufflevector(lo, hi, 0,1,2,3,4,5,6,7,8,9,10,11,12,13,14,15);
}

// ---------------------------------------------------------------------------
// f32 -> f16 elementwise (8 elems/thread); n is a multiple of 2048.
// ---------------------------------------------------------------------------
__global__ __launch_bounds__(256)
void cvt_f32_f16(const float* __restrict__ in, _Float16* __restrict__ out) {
    const size_t i = ((size_t)blockIdx.x * 256 + threadIdx.x) * 8;
    f4 a = *(const f4*)(in + i);
    f4 b = *(const f4*)(in + i + 4);
    h8 o;
#pragma unroll
    for (int j = 0; j < 4; ++j) { o[j] = (_Float16)a[j]; o[4 + j] = (_Float16)b[j]; }
    *(h8*)(out + i) = o;
}

// ---------------------------------------------------------------------------
// GEMM:  C[M,N] = A[M,K] * W[N,K]^T + bias[N],  K = 1024, all f16 inputs.
// MODE 0: result f32 -> outF row-major [M,N].
// MODE 1: result f16 scattered to Q/K/V [B,H,T,64]; Q scaled by 1/8.
// Block tile 128x128, BK=32, 8 waves as 2x4, wave tile 64x32 (4x2 WMMA frags).
// LDS double-buffered, staged with async global->LDS b128 copies.
// ---------------------------------------------------------------------------
template <int MODE>
__global__ __launch_bounds__(256)
void gemm_wmma(const _Float16* __restrict__ Ah, const _Float16* __restrict__ Wh,
               const float* __restrict__ bias, float* __restrict__ outF,
               _Float16* __restrict__ Qo, _Float16* __restrict__ Ko,
               _Float16* __restrict__ Vo, int M, int N) {
    constexpr int K = 1024;
    constexpr int NT = K / 32;            // 32 k-tiles
    __shared__ _Float16 As[2][128][40];   // +8 pad per row
    __shared__ _Float16 Bs[2][128][40];

    const int tid  = threadIdx.x;
    const int wave = tid >> 5, lane = tid & 31;
    const int lr   = lane & 15, hi = lane >> 4;
    const int m0   = blockIdx.y * 128, n0 = blockIdx.x * 128;
    const int wm   = wave >> 2;           // 0..1 -> 64-row half
    const int wn   = wave & 3;            // 0..3 -> 32-col strip

    // staging map: thread t covers row r = t/2, halves [part*16, part*16+16)
    const int sr = tid >> 1, sp = tid & 1;
    const _Float16* Ag = Ah + (size_t)(m0 + sr) * K + sp * 16;
    const _Float16* Bg = Wh + (size_t)(n0 + sr) * K + sp * 16;

    auto stage = [&](int buf, int kb) {
        cp16_g2l(Ag + kb,     &As[buf][sr][sp * 16]);
        cp16_g2l(Ag + kb + 8, &As[buf][sr][sp * 16 + 8]);
        cp16_g2l(Bg + kb,     &Bs[buf][sr][sp * 16]);
        cp16_g2l(Bg + kb + 8, &Bs[buf][sr][sp * 16 + 8]);
    };

    v8f acc[4][2] = {};

    stage(0, 0);                           // prologue: tile 0 in flight
    for (int it = 0; it < NT; ++it) {
        const int cur = it & 1;
        if (it + 1 < NT) {
            stage(1 - cur, (it + 1) * 32); // overlap next tile with compute
#if HAVE_ASYNC
            __builtin_amdgcn_s_wait_asynccnt(4);   // current tile done, next in flight
#endif
        } else {
#if HAVE_ASYNC
            __builtin_amdgcn_s_wait_asynccnt(0);
#endif
        }
        __syncthreads();

        const int ksA = hi ? 16 : 0;  // A frag: lanes 0-15 -> K 0-15     ; 16-31 -> 16-31
        const int ksB = hi ? 16 : 0;  // B frag: lanes 0-15 -> K 0-15     ; 16-31 -> 16-31
        v16h a[4], b[2];
#pragma unroll
        for (int mi = 0; mi < 4; ++mi) {
            const int row = wm * 64 + mi * 16 + lr;
            a[mi] = ld_frag16(&As[cur][row][ksA], &As[cur][row][ksA + 8]);
        }
#pragma unroll
        for (int ni = 0; ni < 2; ++ni) {
            const int col = wn * 32 + ni * 16 + lr;
            b[ni] = ld_frag16(&Bs[cur][col][ksB], &Bs[cur][col][ksB + 8]);
        }
#pragma unroll
        for (int mi = 0; mi < 4; ++mi)
#pragma unroll
            for (int ni = 0; ni < 2; ++ni)
                acc[mi][ni] = __builtin_amdgcn_wmma_f32_16x16x32_f16(
                    false, a[mi], false, b[ni], (short)0, acc[mi][ni], false, false);
        __syncthreads();                  // everyone done reading buf[cur]
    }

    // --- epilogue (D frag: VGPR r -> row r + 8*hi, col = lane%16) ---
#pragma unroll
    for (int mi = 0; mi < 4; ++mi) {
#pragma unroll
        for (int ni = 0; ni < 2; ++ni) {
#pragma unroll
            for (int r = 0; r < 8; ++r) {
                const int gm = m0 + wm * 64 + mi * 16 + r + 8 * hi;
                const int gn = n0 + wn * 32 + ni * 16 + lr;
                float v = acc[mi][ni][r] + bias[gn];
                if (MODE == 0) {
                    outF[(size_t)gm * N + gn] = v;
                } else {
                    const int which = gn >> 10, c = gn & 1023;
                    const int hh = c >> 6, dc = c & 63;
                    const int bb = gm >> 11, t = gm & 2047;
                    const size_t off = (((size_t)(bb * 16 + hh) * 2048) + t) * 64 + dc;
                    if (which == 0)      Qo[off] = (_Float16)(v * 0.125f);  // fold 1/sqrt(64)
                    else if (which == 1) Ko[off] = (_Float16)v;
                    else                 Vo[off] = (_Float16)v;
                }
            }
        }
    }
}

// ---------------------------------------------------------------------------
// Flash attention (causal). Q/K/V f16 [B*H, T, 64], Q pre-scaled by 1/8.
// 256 threads = 8 waves; 128 query rows per block, 16 rows per wave.
// Key tiles of 64; S=Q*K^T and O+=P*V via v_wmma_f32_16x16x32_f16.
// Output f16 [B, T, 1024] for the proj GEMM.
// ---------------------------------------------------------------------------
__global__ __launch_bounds__(256)
void attn_flash(const _Float16* __restrict__ Q, const _Float16* __restrict__ Kg,
                const _Float16* __restrict__ Vg, _Float16* __restrict__ Oh) {
    __shared__ _Float16 Ks[64][72];       // [key][d]
    __shared__ _Float16 Vs[64][72];       // [d][key]  (transposed)
    __shared__ _Float16 Ps[8][16][72];    // per-wave P tile (16 rows x 64 keys)

    const int tid  = threadIdx.x;
    const int wave = tid >> 5, lane = tid & 31;
    const int lr   = lane & 15, hi = lane >> 4;
    const int bh   = blockIdx.y;               // b*16 + h
    const int b    = bh >> 4, h = bh & 15;
    const int qBase = blockIdx.x * 128;
    const int qw    = qBase + wave * 16;       // this wave's first query row
    const size_t headOff = (size_t)bh * 2048 * 64;

    // Q A-fragments (d = 0..31 and 32..63), loaded once
    v16h qa[2];
    {
        const _Float16* qrow = Q + headOff + (size_t)(qw + lr) * 64;
        const int ks = hi ? 16 : 0;
        qa[0] = ld_frag16(qrow + ks,      qrow + ks + 8);
        qa[1] = ld_frag16(qrow + 32 + ks, qrow + 32 + ks + 8);
    }

    v8f o[4] = {};
    float mrow[8], lrow[8];
#pragma unroll
    for (int r = 0; r < 8; ++r) { mrow[r] = NEG_INF; lrow[r] = 0.f; }

    for (int kb = 0; kb <= qBase + 127; kb += 64) {
        // --- cooperative stage: K tile [key][d], V tile transposed [d][key] ---
        {
            const int key = tid >> 2, part = tid & 3;
            const _Float16* ksrc = Kg + headOff + (size_t)(kb + key) * 64 + part * 16;
            *(h8*)&Ks[key][part * 16]     = *(const h8*)(ksrc);
            *(h8*)&Ks[key][part * 16 + 8] = *(const h8*)(ksrc + 8);
            const _Float16* vsrc = Vg + headOff + (size_t)(kb + key) * 64 + part * 16;
#pragma unroll
            for (int i = 0; i < 16; ++i) Vs[part * 16 + i][key] = vsrc[i];
        }
        __syncthreads();

        const int ksB = hi ? 16 : 0;
        // --- S = Q * K^T  (4 key sub-tiles of 16) ---
        v8f s[4];
#pragma unroll
        for (int ni = 0; ni < 4; ++ni) {
            const int col = ni * 16 + lr;
            v16h b0 = ld_frag16(&Ks[col][ksB],      &Ks[col][ksB + 8]);
            v16h b1 = ld_frag16(&Ks[col][32 + ksB], &Ks[col][32 + ksB + 8]);
            v8f z = {};
            z     = __builtin_amdgcn_wmma_f32_16x16x32_f16(false, qa[0], false, b0, (short)0, z, false, false);
            s[ni] = __builtin_amdgcn_wmma_f32_16x16x32_f16(false, qa[1], false, b1, (short)0, z, false, false);
        }
        // --- causal mask (only near/above the diagonal) ---
        if (kb + 63 > qw) {
#pragma unroll
            for (int ni = 0; ni < 4; ++ni) {
                const int col = kb + ni * 16 + lr;
#pragma unroll
                for (int r = 0; r < 8; ++r)
                    if (col > qw + r + 8 * hi) s[ni][r] = NEG_INF;
            }
        }
        // --- online softmax (row r+8*hi lives in VGPR r, keys across 16 lanes) ---
#pragma unroll
        for (int r = 0; r < 8; ++r) {
            float tm = fmaxf(fmaxf(s[0][r], s[1][r]), fmaxf(s[2][r], s[3][r]));
#pragma unroll
            for (int off = 1; off < 16; off <<= 1) tm = fmaxf(tm, __shfl_xor(tm, off, 32));
            const float nm    = fmaxf(mrow[r], tm);        // finite: kb=0 tile always has col<=row
            const float scale = __expf(mrow[r] - nm);
            mrow[r] = nm;
            float rs = 0.f;
#pragma unroll
            for (int ni = 0; ni < 4; ++ni) {
                const float p = __expf(s[ni][r] - nm);
                s[ni][r] = p;
                rs += p;
            }
#pragma unroll
            for (int off = 1; off < 16; off <<= 1) rs += __shfl_xor(rs, off, 32);
            lrow[r] = lrow[r] * scale + rs;
#pragma unroll
            for (int ni = 0; ni < 4; ++ni) o[ni][r] *= scale;
            // re-layout P: D-fragment -> LDS (read back below as A-fragment)
#pragma unroll
            for (int ni = 0; ni < 4; ++ni)
                Ps[wave][r + 8 * hi][ni * 16 + lr] = (_Float16)s[ni][r];
        }
        asm volatile("" ::: "memory");   // keep LDS P stores before the reloads
        // --- O += P * V ---
        v16h pa[2];
        {
            const _Float16* prow = &Ps[wave][lr][0];
            const int ks = hi ? 16 : 0;
            pa[0] = ld_frag16(prow + ks,      prow + ks + 8);
            pa[1] = ld_frag16(prow + 32 + ks, prow + 32 + ks + 8);
        }
#pragma unroll
        for (int ni = 0; ni < 4; ++ni) {
            const int col = ni * 16 + lr;     // output d column
            v16h v0 = ld_frag16(&Vs[col][ksB],      &Vs[col][ksB + 8]);
            v16h v1 = ld_frag16(&Vs[col][32 + ksB], &Vs[col][32 + ksB + 8]);
            o[ni] = __builtin_amdgcn_wmma_f32_16x16x32_f16(false, pa[0], false, v0, (short)0, o[ni], false, false);
            o[ni] = __builtin_amdgcn_wmma_f32_16x16x32_f16(false, pa[1], false, v1, (short)0, o[ni], false, false);
        }
        __syncthreads();
    }

    // --- normalize and store f16 [B, T, H*64] ---
#pragma unroll
    for (int ni = 0; ni < 4; ++ni) {
#pragma unroll
        for (int r = 0; r < 8; ++r) {
            const int t   = qw + r + 8 * hi;
            const int col = h * 64 + ni * 16 + lr;
            Oh[((size_t)(b * 2048 + t)) * 1024 + col] = (_Float16)(o[ni][r] / lrow[r]);
        }
    }
}

// ---------------------------------------------------------------------------
extern "C" void kernel_launch(void* const* d_in, const int* in_sizes, int n_in,
                              void* d_out, int out_size, void* d_ws, size_t ws_size,
                              hipStream_t stream) {
    const float* x     = (const float*)d_in[0];   // [4,2048,1024]
    const float* Wqkv  = (const float*)d_in[1];   // [3072,1024]
    const float* bqkv  = (const float*)d_in[2];   // [3072]
    const float* Wproj = (const float*)d_in[3];   // [1024,1024]
    const float* bproj = (const float*)d_in[4];   // [1024]
    float* out = (float*)d_out;                   // [4,2048,1024]

    constexpr size_t NQKV = (size_t)4 * 16 * 2048 * 64;        // 8.4M elems per tensor
    constexpr size_t NX   = (size_t)4 * 2048 * 1024;           // 8.4M
    constexpr size_t NWQ  = (size_t)3072 * 1024;               // 3.1M
    constexpr size_t NWP  = (size_t)1024 * 1024;               // 1.0M

    char* ws = (char*)d_ws;
    _Float16* Q    = (_Float16*)(ws);
    _Float16* K    = (_Float16*)(ws + 1 * NQKV * 2);
    _Float16* V    = (_Float16*)(ws + 2 * NQKV * 2);
    _Float16* xh   = (_Float16*)(ws + 3 * NQKV * 2);           // aliased with attO
    _Float16* attO = xh;                                       // x dead after QKV GEMM
    _Float16* Wqh  = (_Float16*)(ws + 3 * NQKV * 2 + NX * 2);
    _Float16* Wph  = (_Float16*)(ws + 3 * NQKV * 2 + NX * 2 + NWQ * 2);

    // 0) one-pass f32 -> f16 conversions (bandwidth noise vs. the GEMM FLOPs)
    cvt_f32_f16<<<dim3(NX  / 2048), 256, 0, stream>>>(x,     xh);
    cvt_f32_f16<<<dim3(NWQ / 2048), 256, 0, stream>>>(Wqkv,  Wqh);
    cvt_f32_f16<<<dim3(NWP / 2048), 256, 0, stream>>>(Wproj, Wph);

    // 1) QKV GEMM: xh * Wqkv^T + b -> f16 Q(scaled)/K/V [B,H,T,64]
    gemm_wmma<1><<<dim3(3072 / 128, 8192 / 128), 256, 0, stream>>>(
        xh, Wqh, bqkv, nullptr, Q, K, V, 8192, 3072);

    // 2) causal flash attention -> f16 [B,T,1024]
    attn_flash<<<dim3(2048 / 128, 4 * 16), 256, 0, stream>>>(Q, K, V, attO);

    // 3) proj GEMM: attO * Wproj^T + b -> f32 out
    gemm_wmma<0><<<dim3(1024 / 128, 8192 / 128), 256, 0, stream>>>(
        attO, Wph, bproj, out, nullptr, nullptr, nullptr, 8192, 1024);
}
